// ContrastiveLoss_45311904972892
// MI455X (gfx1250) — matrix operand order repaired and checked by reading, not compiled, not run on hardware
//
#include <hip/hip_runtime.h>
#include <hip/hip_bf16.h>

// ---------------------------------------------------------------------------
// ContrastiveLoss on MI455X (gfx1250):
//   scores = feature @ anchor^T   (16384 x 2048) x (2048 x 1000)
//   loss   = mean_b [ logsumexp_c(scores[b,:]) - scores[b, target[b]] ]
// v3: 32 rows/block (halves L2 anchor traffic), 2x2 register tiling
//     (4 WMMAs share 2 A + 2 B fragments), branchless softmax fold.
// ---------------------------------------------------------------------------

#define B_ROWS   16384
#define N_CLS    1000
#define K_FEAT   2048
#define WAVES    4
#define BLOCK_T  (WAVES * 32)
#define ROWS_BLK 32            // rows of B handled per workgroup (2 M-tiles)
#define N_PAIRS  32            // 32 pairs x 32 classes = 1024 >= 1000

typedef __attribute__((ext_vector_type(16))) __bf16 v16bf;
typedef __attribute__((ext_vector_type(8)))  float  v8f;

static __device__ __forceinline__ unsigned short f2bf(float x) {
    unsigned u = __float_as_uint(x);
    unsigned r = (u + 0x7FFFu + ((u >> 16) & 1u)) >> 16;   // RNE to bf16
    return (unsigned short)r;
}
static __device__ __forceinline__ unsigned pack2bf(float a, float b) {
    return (unsigned)f2bf(a) | ((unsigned)f2bf(b) << 16);
}

// ---- Prologue: anchor fp32 -> bf16 (1000 x 2048 = 4 MB), once per launch ---
__global__ __launch_bounds__(256)
void convert_anchor_kernel(const float* __restrict__ anchor,
                           unsigned short* __restrict__ abf) {
    const int i = (blockIdx.x * 256 + threadIdx.x) * 4;
    if (i + 3 < N_CLS * K_FEAT) {
        const float4 f = *(const float4*)(anchor + i);
        *(uint2*)&abf[i] = make_uint2(pack2bf(f.x, f.y), pack2bf(f.z, f.w));
    }
}

// Branchless online-softmax fold of one finished 16x16 score tile.
static __device__ __forceinline__ void fold_tile(
        const v8f& a, int n, int mt, int hi,
        float (&m_run)[2][8], float (&s_run)[2][8], float (&t_run)[2][8],
        const int* sT) {
    const bool valid = (n < N_CLS);
#pragma unroll
    for (int v = 0; v < 8; ++v) {
        const float sc = valid ? a[v] : -1e30f;
        const float mn = fmaxf(m_run[mt][v], sc);
        s_run[mt][v] = s_run[mt][v] * __expf(m_run[mt][v] - mn)
                     + (valid ? __expf(sc - mn) : 0.f);
        m_run[mt][v] = mn;
        const int r = mt * 16 + v + 8 * hi;            // row within block tile
        t_run[mt][v] += (valid && n == sT[r]) ? a[v] : 0.f;
    }
}

__global__ __launch_bounds__(BLOCK_T)
void contrastive_fused_kernel(const float* __restrict__ feature,
                              const unsigned short* __restrict__ abf,
                              const int*   __restrict__ target,
                              float*       __restrict__ partial) {
    __shared__ unsigned short sA[ROWS_BLK * K_FEAT];   // 32-row bf16 tile (128 KB)
    __shared__ float sRed[WAVES][ROWS_BLK][3];
    __shared__ int   sT[ROWS_BLK];

    const int tid  = threadIdx.x;
    const int wv   = tid >> 5;
    const int lane = tid & 31;
    const int hi   = lane >> 4;        // 0: lanes 0-15, 1: lanes 16-31
    const int lo   = lane & 15;
    const int r0   = blockIdx.x * ROWS_BLK;

    // ---- Stage 32 x 2048 fp32 feature rows into LDS as bf16 (one time) -----
    {
        const float* frow = feature + (size_t)r0 * K_FEAT;
        for (int i = tid * 4; i < ROWS_BLK * K_FEAT; i += BLOCK_T * 4) {
            const float4 f = *(const float4*)(frow + i);
            *(uint2*)&sA[i] = make_uint2(pack2bf(f.x, f.y), pack2bf(f.z, f.w));
        }
        if (tid < ROWS_BLK) sT[tid] = target[r0 + tid];
    }
    __syncthreads();

    // ---- Per-lane online-softmax state: 2 M-tiles x 8 rows ------------------
    float m_run[2][8], s_run[2][8], t_run[2][8];
#pragma unroll
    for (int mt = 0; mt < 2; ++mt)
#pragma unroll
        for (int v = 0; v < 8; ++v) {
            m_run[mt][v] = -1e30f; s_run[mt][v] = 0.f; t_run[mt][v] = 0.f;
        }

    const unsigned short* sArow0 = &sA[lo * K_FEAT + 8 * hi];          // rows 0-15
    const unsigned short* sArow1 = &sA[(16 + lo) * K_FEAT + 8 * hi];   // rows 16-31

    // ---- Loop over 32-class tile pairs; waves interleave pairs --------------
    for (int p = wv; p < N_PAIRS; p += WAVES) {
        const int n0 = p * 32;
        const int na = n0 + lo;            // class column, tile 0 of pair
        const int nb = na + 16;            // class column, tile 1 of pair
        const int nca = na < N_CLS ? na : (N_CLS - 1);
        const int ncb = nb < N_CLS ? nb : (N_CLS - 1);
        const unsigned short* ap0 = abf + (size_t)nca * K_FEAT + 8 * hi;
        const unsigned short* ap1 = abf + (size_t)ncb * K_FEAT + 8 * hi;

        v8f acc00 = {}, acc01 = {}, acc10 = {}, acc11 = {};
#pragma unroll 2
        for (int k0 = 0; k0 < K_FEAT; k0 += 32) {
            // Two A fragments (rows 0-15 / 16-31), two B fragments (2 tiles)
            union { v16bf v; uint4 q[2]; } A0, A1, B0, B1;
            A0.q[0] = *(const uint4*)(sArow0 + k0);
            A0.q[1] = *(const uint4*)(sArow0 + k0 + 16);
            A1.q[0] = *(const uint4*)(sArow1 + k0);
            A1.q[1] = *(const uint4*)(sArow1 + k0 + 16);
            B0.q[0] = *(const uint4*)(ap0 + k0);
            B0.q[1] = *(const uint4*)(ap0 + k0 + 16);
            B1.q[0] = *(const uint4*)(ap1 + k0);
            B1.q[1] = *(const uint4*)(ap1 + k0 + 16);

            acc00 = __builtin_amdgcn_wmma_f32_16x16x32_bf16(
                        false, A0.v, false, B0.v, (short)0, acc00, false, false);
            acc01 = __builtin_amdgcn_wmma_f32_16x16x32_bf16(
                        false, A0.v, false, B1.v, (short)0, acc01, false, false);
            acc10 = __builtin_amdgcn_wmma_f32_16x16x32_bf16(
                        false, A1.v, false, B0.v, (short)0, acc10, false, false);
            acc11 = __builtin_amdgcn_wmma_f32_16x16x32_bf16(
                        false, A1.v, false, B1.v, (short)0, acc11, false, false);
        }

        fold_tile(acc00, na, 0, hi, m_run, s_run, t_run, sT);
        fold_tile(acc01, nb, 0, hi, m_run, s_run, t_run, sT);
        fold_tile(acc10, na, 1, hi, m_run, s_run, t_run, sT);
        fold_tile(acc11, nb, 1, hi, m_run, s_run, t_run, sT);
    }

    // ---- Cross-lane reduce (16 lanes per half hold one row's 16 classes) ---
#pragma unroll
    for (int mt = 0; mt < 2; ++mt)
#pragma unroll
        for (int v = 0; v < 8; ++v) {
            float m = m_run[mt][v], s = s_run[mt][v], tt = t_run[mt][v];
#pragma unroll
            for (int mask = 1; mask <= 8; mask <<= 1) {
                const float m2 = __shfl_xor(m, mask, 32);
                const float s2 = __shfl_xor(s, mask, 32);
                const float t2 = __shfl_xor(tt, mask, 32);
                const float mn = fmaxf(m, m2);
                s  = s * __expf(m - mn) + s2 * __expf(m2 - mn);
                m  = mn;
                tt = tt + t2;
            }
            if (lo == 0) {                 // lane 0 -> rows v, lane 16 -> v+8
                const int r = mt * 16 + v + 8 * hi;
                sRed[wv][r][0] = m;
                sRed[wv][r][1] = s;
                sRed[wv][r][2] = tt;
            }
        }
    __syncthreads();

    // ---- Combine 4 waves, emit per-block NLL partial sum --------------------
    if (tid == 0) {
        float bsum = 0.f;
        for (int r = 0; r < ROWS_BLK; ++r) {
            float M = -1e30f;
            for (int w = 0; w < WAVES; ++w) M = fmaxf(M, sRed[w][r][0]);
            float S = 0.f, T = 0.f;
            for (int w = 0; w < WAVES; ++w) {
                S += sRed[w][r][1] * __expf(sRed[w][r][0] - M);
                T += sRed[w][r][2];
            }
            bsum += M + __logf(S) - T;     // logsumexp - target score
        }
        partial[blockIdx.x] = bsum;
    }
}

__global__ __launch_bounds__(128)
void reduce_mean_kernel(const float* __restrict__ partial, float* __restrict__ out,
                        int n, float inv) {
    __shared__ float sm[128];
    float s = 0.f;
    for (int i = threadIdx.x; i < n; i += 128) s += partial[i];
    sm[threadIdx.x] = s;
    __syncthreads();
    for (int off = 64; off > 0; off >>= 1) {
        if (threadIdx.x < off) sm[threadIdx.x] += sm[threadIdx.x + off];
        __syncthreads();
    }
    if (threadIdx.x == 0) out[0] = sm[0] * inv;
}

extern "C" void kernel_launch(void* const* d_in, const int* in_sizes, int n_in,
                              void* d_out, int out_size, void* d_ws, size_t ws_size,
                              hipStream_t stream) {
    const float* feature = (const float*)d_in[0];
    const float* anchor  = (const float*)d_in[1];
    const int*   target  = (const int*)d_in[2];
    float* out = (float*)d_out;

    // Workspace layout: [0, 4KB) block partials; [4KB, 4KB+4MB) bf16 anchor.
    float*          partial = (float*)d_ws;
    unsigned short* abf     = (unsigned short*)((char*)d_ws + 4096);

    const int cvt_elems  = N_CLS * K_FEAT;                 // 2,048,000
    const int cvt_blocks = (cvt_elems / 4 + 255) / 256;    // 4 elems/thread
    convert_anchor_kernel<<<cvt_blocks, 256, 0, stream>>>(anchor, abf);

    const int nblocks = B_ROWS / ROWS_BLK;                 // 512
    contrastive_fused_kernel<<<nblocks, BLOCK_T, 0, stream>>>(feature, abf, target, partial);
    reduce_mean_kernel<<<1, 128, 0, stream>>>(partial, out, nblocks, 1.0f / (float)B_ROWS);
}
